// GATConv_15006615733819
// MI455X (gfx1250) — matrix operand order repaired and checked by reading, not compiled, not run on hardware
//
#include <hip/hip_runtime.h>
#include <hip/hip_bf16.h>
#include <math.h>

// ---------------------------------------------------------------------------
// GATConv on MI455X (gfx1250):
//   Phase 1: xp = x @ W  via v_wmma_f32_16x16x32_bf16 (fp32 accum)
//   Phase 2: per-node head scores s_i, s_j
//   Phase 3: per-edge exp(leaky_relu) scatter-add into node accumulator
//   Phase 4: per-head denominator (block-reduced atomics)
//   Phase 5: init 409.6MB output with bias (float4 streaming stores)
//   Phase 6: scatter-add xp[ej]*alpha into out[ei] (global_atomic_add_f32)
// ---------------------------------------------------------------------------

typedef __attribute__((ext_vector_type(16))) __bf16 v16bf;
typedef __attribute__((ext_vector_type(8)))  float  v8f;

#define HEADS 8
#define CPH   16
#define HC    128   // HEADS * CPH

// ---------------- Phase 1: GEMM xp[N,128] = x[N,128] @ W[128,128] ----------
// One wave per 16x16 output tile; 8 waves/block -> one 16-row strip per block.
__global__ __launch_bounds__(256) void gat_gemm_xp(const float* __restrict__ x,
                                                   const float* __restrict__ W,
                                                   float* __restrict__ xp) {
  const int lane    = threadIdx.x & 31;
  const int wave    = threadIdx.x >> 5;     // column tile 0..7
  const int rowTile = blockIdx.x;           // 0..N/16-1
  const int m       = lane & 15;
  const int hi      = lane >> 4;            // 0 or 1 (half-wave)

  const int arow = rowTile * 16 + m;        // A row this lane owns
  const int bcol = wave * 16 + m;           // B column this lane owns

  v8f acc = {};
#pragma unroll
  for (int kk = 0; kk < 4; ++kk) {
    const int kbase = kk * 32;
    // A (16x32 bf16): lanes 0-15 hold K = kbase+0..15; lanes 16-31 kbase+16..31.
    v16bf a;
    const float* xr = x + arow * HC + kbase + hi * 16;
#pragma unroll
    for (int t = 0; t < 8; ++t) a[t]     = (__bf16)xr[t];
#pragma unroll
    for (int t = 0; t < 8; ++t) a[8 + t] = (__bf16)xr[8 + t];
    // B (32x16 bf16): lanes 0-15 -> K = kbase+0..15, lanes 16-31 -> kbase+16..31.
    v16bf b;
    const float* wc = W + (kbase + hi * 16) * HC + bcol;
#pragma unroll
    for (int t = 0; t < 16; ++t) b[t] = (__bf16)wc[t * HC];

    acc = __builtin_amdgcn_wmma_f32_16x16x32_bf16(
        /*neg_a=*/false, a, /*neg_b=*/false, b,
        /*c_mod=*/(short)0, acc, /*reuse_a=*/false, /*reuse_b=*/false);
  }
  // C/D layout: element v -> M = v + hi*8, N = m
  float* outp = xp + (rowTile * 16 + hi * 8) * HC + wave * 16 + m;
#pragma unroll
  for (int v = 0; v < 8; ++v) outp[v * HC] = acc[v];
}

// ---------------- Phase 2: s_i[n,h], s_j[n,h] ------------------------------
__global__ void gat_scores(const float* __restrict__ xp,
                           const float* __restrict__ att,  // [2C, H] row-major
                           float* __restrict__ s_i, float* __restrict__ s_j,
                           int N) {
  int idx = blockIdx.x * blockDim.x + threadIdx.x;
  if (idx >= N * HEADS) return;
  const int n = idx >> 3, h = idx & 7;
  const float* xph = xp + n * HC + h * CPH;
  float si = 0.f, sj = 0.f;
#pragma unroll
  for (int c = 0; c < CPH; ++c) {
    const float v = xph[c];
    si += v * att[c * HEADS + h];
    sj += v * att[(CPH + c) * HEADS + h];
  }
  s_i[idx] = si;
  s_j[idx] = sj;
}

// ---------------- zero helper ----------------------------------------------
__global__ void gat_zero(float* __restrict__ p, int n) {
  int idx = blockIdx.x * blockDim.x + threadIdx.x;
  if (idx < n) p[idx] = 0.f;
}

// ---------------- Phase 3: alphaAcc[ei,h] += exp(leaky_relu(...)) ----------
__global__ void gat_edge_exp(const int* __restrict__ ei, const int* __restrict__ ej,
                             const float* __restrict__ s_i, const float* __restrict__ s_j,
                             float* __restrict__ alphaAcc, int E) {
  int idx = blockIdx.x * blockDim.x + threadIdx.x;
  if (idx >= E * HEADS) return;
  const int e = idx >> 3, h = idx & 7;
  const int i = ei[e], j = ej[e];
  float v = s_i[i * HEADS + h] + s_j[j * HEADS + h];
  v = v > 0.f ? v : 0.2f * v;              // leaky_relu(0.2)
  atomicAdd(&alphaAcc[i * HEADS + h], __expf(v));
}

// ---------------- Phase 4: denom[h] = sum_e alphaAcc[ej[e], h] -------------
__global__ __launch_bounds__(256) void gat_denom(const int* __restrict__ ej,
                                                 const float* __restrict__ alphaAcc,
                                                 float* __restrict__ denom, int E) {
  __shared__ float sh[HEADS];
  if (threadIdx.x < HEADS) sh[threadIdx.x] = 0.f;
  __syncthreads();
  float acc[HEADS];
#pragma unroll
  for (int h = 0; h < HEADS; ++h) acc[h] = 0.f;
  for (int e = blockIdx.x * blockDim.x + threadIdx.x; e < E;
       e += gridDim.x * blockDim.x) {
    const int j = ej[e];
#pragma unroll
    for (int h = 0; h < HEADS; ++h) acc[h] += alphaAcc[j * HEADS + h];
  }
#pragma unroll
  for (int h = 0; h < HEADS; ++h) atomicAdd(&sh[h], acc[h]);
  __syncthreads();
  if (threadIdx.x < HEADS) atomicAdd(&denom[threadIdx.x], sh[threadIdx.x]);
}

// ---------------- Phase 5: out[r,:] = bias (float4 streaming) --------------
__global__ void gat_init_out(float* __restrict__ out, const float* __restrict__ bias,
                             int nquads) {   // nquads = E * 32
  int idx = blockIdx.x * blockDim.x + threadIdx.x;
  if (idx >= nquads) return;
  const int q = idx & 31;                   // 32 float4's per 128-float row
  ((float4*)out)[idx] = ((const float4*)bias)[q];
}

// ---------------- Phase 6: out[ei] += xp[ej] * alpha -----------------------
__global__ void gat_scatter_out(const int* __restrict__ ei, const int* __restrict__ ej,
                                const float* __restrict__ xp,
                                const float* __restrict__ alphaAcc,
                                const float* __restrict__ denom,
                                float* __restrict__ out, int E) {
  int idx = blockIdx.x * blockDim.x + threadIdx.x;   // E*32 threads
  if (idx >= E * 32) return;
  const int q = idx & 31;                   // quad 0..31 within row
  const int e = idx >> 5;
  const int i = ei[e], j = ej[e];
  const int h = q >> 2;                     // 4 quads per head (16 channels)
  const float scale = alphaAcc[j * HEADS + h] / (denom[h] + 1e-16f);
  const float4 v = *(const float4*)(xp + j * HC + q * 4);
  float* dst = out + (size_t)i * HC + q * 4;
  atomicAdd(dst + 0, v.x * scale);
  atomicAdd(dst + 1, v.y * scale);
  atomicAdd(dst + 2, v.z * scale);
  atomicAdd(dst + 3, v.w * scale);
}

// ---------------------------------------------------------------------------
extern "C" void kernel_launch(void* const* d_in, const int* in_sizes, int n_in,
                              void* d_out, int out_size, void* d_ws, size_t ws_size,
                              hipStream_t stream) {
  const float* x    = (const float*)d_in[0];
  const int*   eIdx = (const int*)d_in[1];     // [2, E]
  const float* W    = (const float*)d_in[2];
  const float* att  = (const float*)d_in[3];
  const float* bias = (const float*)d_in[4];
  float* out = (float*)d_out;

  const int N = in_sizes[0] / HC;              // 50000
  const int E = in_sizes[1] / 2;               // 800000
  const int* ei = eIdx;
  const int* ej = eIdx + E;

  // Workspace partition (bytes, 256-aligned)
  char* ws = (char*)d_ws;
  size_t off = 0;
  auto alloc = [&](size_t bytes) { char* p = ws + off; off = (off + bytes + 255) & ~(size_t)255; return p; };
  float* xp       = (float*)alloc((size_t)N * HC * sizeof(float));      // 25.6 MB
  float* s_i      = (float*)alloc((size_t)N * HEADS * sizeof(float));   // 1.6 MB
  float* s_j      = (float*)alloc((size_t)N * HEADS * sizeof(float));   // 1.6 MB
  float* alphaAcc = (float*)alloc((size_t)N * HEADS * sizeof(float));   // 1.6 MB
  float* denom    = (float*)alloc(HEADS * sizeof(float));
  (void)ws_size; (void)n_in; (void)out_size;

  // Phase 1: GEMM (WMMA bf16 -> f32)
  gat_gemm_xp<<<N / 16, 256, 0, stream>>>(x, W, xp);

  // Phase 2: head scores
  gat_scores<<<(N * HEADS + 255) / 256, 256, 0, stream>>>(xp, att, s_i, s_j, N);

  // Zero accumulators (must be re-zeroed every call for determinism)
  gat_zero<<<(N * HEADS + 255) / 256, 256, 0, stream>>>(alphaAcc, N * HEADS);
  gat_zero<<<1, HEADS, 0, stream>>>(denom, HEADS);

  // Phase 3: edge softmax numerator scatter
  gat_edge_exp<<<(E * HEADS + 255) / 256, 256, 0, stream>>>(ei, ej, s_i, s_j, alphaAcc, E);

  // Phase 4: global per-head denominator
  gat_denom<<<512, 256, 0, stream>>>(ej, alphaAcc, denom, E);

  // Phase 5: init full E x 128 output with bias
  const int nquads = E * 32;
  gat_init_out<<<(nquads + 255) / 256, 256, 0, stream>>>(out, bias, nquads);

  // Phase 6: weighted scatter into output
  gat_scatter_out<<<(E * 32 + 255) / 256, 256, 0, stream>>>(ei, ej, xp, alphaAcc, denom, out, E);
}
